// biLSTM_CNN_CRF_49709951484244
// MI455X (gfx1250) — matrix operand order repaired and checked by reading, not compiled, not run on hardware
//
#include <hip/hip_runtime.h>
#include <hip/hip_bf16.h>

typedef __attribute__((ext_vector_type(16))) _Float16 v16h;
typedef __attribute__((ext_vector_type(8)))  float    v8f;

#define CRF_B 1024
#define CRF_T 512
#define CRF_K 48
#define CRF_NEG (-10000.0f)
#define CRF_START (CRF_K - 2)
#define CRF_STOP  (CRF_K - 1)
#define NEGBIG (-3.0e38f)

// ---------------------------------------------------------------------------
// Forward algorithm via WMMA.
// One wave (32 threads) handles 16 batch columns for all T steps.
//   alpha'[next,b] = emit[next,b] + m[b] + log( ExpT[next,prev] @ exp(alpha-m)[prev,b] )
// ExpT = exp(masked trans) is constant -> prebuilt A fragments (16x32 f16 layout).
// exp(alpha-m) is staged per step in LDS in the B (32x16, K-major) layout.
// ---------------------------------------------------------------------------
__global__ __launch_bounds__(32) void crf_forward_wmma(
    const float* __restrict__ feats,
    const float* __restrict__ trans,
    float* __restrict__ logZ)
{
    __shared__ float     mT[CRF_K][CRF_K];   // masked transitions [next][prev]
    __shared__ _Float16  expsT[16][64];      // [batch col][prev row], rows 48..63 stay 0

    const int lane = threadIdx.x;
    const int mloc = lane & 15;   // batch column within tile (C/D: N = lane&15)
    const int hi   = lane >> 4;   // lane-half selector

    // masked transitions into LDS
    for (int i = lane; i < CRF_K * CRF_K; i += 32) {
        int nx = i / CRF_K, pv = i % CRF_K;
        float v = trans[i];
        if (nx == CRF_START) v = CRF_NEG;
        if (pv == CRF_STOP)  v = CRF_NEG;
        mT[nx][pv] = v;
    }
    // zero the staging buffer (pad rows 48..63 remain zero forever)
    for (int i = lane; i < 16 * 64; i += 32)
        ((_Float16*)expsT)[i] = (_Float16)0.0f;
    __syncthreads();

    // Build A fragments of ExpT: a[mt][kc] covers next rows mt*16..+15, prev cols kc*32..+31.
    // A k order used here (contiguous, same as B): lane m = lane&15; element h -> K = (lane>>4)*16 + h.
    v16h a[3][2];
#pragma unroll
    for (int mt = 0; mt < 3; ++mt)
#pragma unroll
        for (int kc = 0; kc < 2; ++kc)
#pragma unroll
            for (int h = 0; h < 16; ++h) {
                int kloc = hi * 16 + h;
                int prev = kc * 32 + kloc;
                int next = mt * 16 + mloc;
                float val = (prev < CRF_K) ? __expf(mT[next][prev]) : 0.0f;
                a[mt][kc][h] = (_Float16)val;
            }

    // Init: alpha0 = NEG except START=0  ->  exp(alpha0 - 0) = one-hot(START), m=0
    if (lane < 16) expsT[lane][CRF_START] = (_Float16)1.0f;
    __syncthreads();

    const int btile = blockIdx.x;                 // 0..63
    const float* fbase = feats + (size_t)(btile * 16) * CRF_T * CRF_K;
    const float* fcol  = fbase + (size_t)mloc * CRF_T * CRF_K;  // this lane's batch column

    float mprev = 0.0f;        // running per-column max (absolute scale)
    float alpha[3][8];         // this lane's 24 alpha entries (C layout rows)

    for (int t = 0; t < CRF_T; ++t) {
        // B fragments (32x16 f16, K-major): lane reads 16 consecutive prev rows of col mloc
        v16h bfrag[2];
#pragma unroll
        for (int kc = 0; kc < 2; ++kc) {
            int rbase = kc * 32 + hi * 16;
#pragma unroll
            for (int h = 0; h < 16; ++h)
                bfrag[kc][h] = expsT[mloc][rbase + h];
        }

        // Y[48x16] = ExpT @ X, 3 M-tiles x 2 K-chunks
        v8f c[3];
#pragma unroll
        for (int mt = 0; mt < 3; ++mt) {
            v8f acc = {};
            acc = __builtin_amdgcn_wmma_f32_16x16x32_f16(false, a[mt][0], false, bfrag[0],
                                                         (short)0, acc, false, false);
            acc = __builtin_amdgcn_wmma_f32_16x16x32_f16(false, a[mt][1], false, bfrag[1],
                                                         (short)0, acc, false, false);
            c[mt] = acc;
        }

        // alpha = emit + mprev + log(Y); track per-column max
        const float* fptr = fcol + (size_t)t * CRF_K;
        float lmax = NEGBIG;
#pragma unroll
        for (int mt = 0; mt < 3; ++mt)
#pragma unroll
            for (int r = 0; r < 8; ++r) {
                int mrow = mt * 16 + hi * 8 + r;          // C layout: M = vgpr + 8*hi
                float av = fptr[mrow] + mprev + __logf(c[mt][r]);
                alpha[mt][r] = av;
                lmax = fmaxf(lmax, av);
            }
        float mcol = fmaxf(lmax, __shfl_xor(lmax, 16, 32));

        // stage exp(alpha - mcol) as f16 into [col][row] buffer
        __syncthreads();
#pragma unroll
        for (int mt = 0; mt < 3; ++mt) {
            int rbase = mt * 16 + hi * 8;
#pragma unroll
            for (int r = 0; r < 8; ++r)
                expsT[mloc][rbase + r] = (_Float16)__expf(alpha[mt][r] - mcol);
        }
        __syncthreads();
        mprev = mcol;
    }

    // logZ = LSE_k( alpha_T[k] + trans[STOP][k] )
    float term[3][8];
    float tmax = NEGBIG;
#pragma unroll
    for (int mt = 0; mt < 3; ++mt)
#pragma unroll
        for (int r = 0; r < 8; ++r) {
            int mrow = mt * 16 + hi * 8 + r;
            float tv = alpha[mt][r] + mT[CRF_STOP][mrow];
            term[mt][r] = tv;
            tmax = fmaxf(tmax, tv);
        }
    tmax = fmaxf(tmax, __shfl_xor(tmax, 16, 32));
    float s = 0.0f;
#pragma unroll
    for (int mt = 0; mt < 3; ++mt)
#pragma unroll
        for (int r = 0; r < 8; ++r)
            s += __expf(term[mt][r] - tmax);
    s += __shfl_xor(s, 16, 32);
    if (hi == 0)
        logZ[btile * 16 + mloc] = tmax + __logf(s);
}

// ---------------------------------------------------------------------------
// Viterbi: 64 threads per batch element (48 active "next" states), 4 per block.
// Backpointers as u8 in workspace; backtrace by one lane per batch element.
// ---------------------------------------------------------------------------
__global__ __launch_bounds__(256) void crf_viterbi(
    const float* __restrict__ feats,
    const float* __restrict__ trans,
    unsigned char* __restrict__ bp,
    float* __restrict__ out)      // out: [nll B][path_score B][paths B*T as int32]
{
    __shared__ float mT[CRF_K][CRF_K + 1];   // pad to 49 -> conflict-free strided reads
    __shared__ float v[4][CRF_K];
    __shared__ float term[4][CRF_K];

    const int tid = threadIdx.x;
    const int be  = tid >> 6;          // batch element within block: 0..3
    const int nxt = tid & 63;          // candidate next state (active if < K)
    const int b   = blockIdx.x * 4 + be;

    for (int i = tid; i < CRF_K * CRF_K; i += 256) {
        int nx = i / CRF_K, pv = i % CRF_K;
        float val = trans[i];
        if (nx == CRF_START) val = CRF_NEG;
        if (pv == CRF_STOP)  val = CRF_NEG;
        mT[nx][pv] = val;
    }
    if (nxt < CRF_K) v[be][nxt] = (nxt == CRF_START) ? 0.0f : CRF_NEG;
    __syncthreads();

    const float* fb = feats + (size_t)b * CRF_T * CRF_K;
    unsigned char* bpb = bp + (size_t)b * CRF_T * CRF_K;

    for (int t = 0; t < CRF_T; ++t) {
        float best = NEGBIG;
        int   arg  = 0;
        if (nxt < CRF_K) {
#pragma unroll 8
            for (int pv = 0; pv < CRF_K; ++pv) {
                float s = v[be][pv] + mT[nxt][pv];
                if (s > best) { best = s; arg = pv; }   // first-max = jnp.argmax semantics
            }
            best += fb[(size_t)t * CRF_K + nxt];
            bpb[(size_t)t * CRF_K + nxt] = (unsigned char)arg;
        }
        __syncthreads();
        if (nxt < CRF_K) v[be][nxt] = best;
        __syncthreads();
    }

    if (nxt < CRF_K) term[be][nxt] = v[be][nxt] + mT[CRF_STOP][nxt];
    __syncthreads();

    if (nxt == 0) {
        float bestv = term[be][0];
        int   bl    = 0;
        for (int k = 1; k < CRF_K; ++k)
            if (term[be][k] > bestv) { bestv = term[be][k]; bl = k; }
        out[CRF_B + b] = bestv;                          // path_score
        int* paths = (int*)(out + 2 * CRF_B) + (size_t)b * CRF_T;
        int cur = bl;
        for (int t = CRF_T - 1; t >= 0; --t) {
            paths[t] = cur;
            cur = (int)bpb[(size_t)t * CRF_K + cur];
        }
    }
}

// ---------------------------------------------------------------------------
// Gold score + nll = logZ - gold. One 128-thread block per batch element.
// ---------------------------------------------------------------------------
__global__ __launch_bounds__(128) void crf_gold_nll(
    const float* __restrict__ feats,
    const float* __restrict__ trans,
    const int* __restrict__ tags,
    const float* __restrict__ logZ,
    float* __restrict__ nll)
{
    __shared__ float mT[CRF_K][CRF_K];
    __shared__ float red[128];

    const int tid = threadIdx.x;
    const int b   = blockIdx.x;

    for (int i = tid; i < CRF_K * CRF_K; i += 128) {
        int nx = i / CRF_K, pv = i % CRF_K;
        float val = trans[i];
        if (nx == CRF_START) val = CRF_NEG;
        if (pv == CRF_STOP)  val = CRF_NEG;
        mT[nx][pv] = val;
    }
    __syncthreads();

    const float* fb = feats + (size_t)b * CRF_T * CRF_K;
    const int*   tg = tags + (size_t)b * CRF_T;

    float s = 0.0f;
    for (int t = tid; t < CRF_T; t += 128) {
        int tag  = tg[t];
        int prev = (t == 0) ? CRF_START : tg[t - 1];
        s += fb[(size_t)t * CRF_K + tag] + mT[tag][prev];
    }
    red[tid] = s;
    __syncthreads();
#pragma unroll
    for (int off = 64; off > 0; off >>= 1) {
        if (tid < off) red[tid] += red[tid + off];
        __syncthreads();
    }
    if (tid == 0) {
        float gold = red[0] + mT[CRF_STOP][tg[CRF_T - 1]];
        nll[b] = logZ[b] - gold;
    }
}

// ---------------------------------------------------------------------------
extern "C" void kernel_launch(void* const* d_in, const int* in_sizes, int n_in,
                              void* d_out, int out_size, void* d_ws, size_t ws_size,
                              hipStream_t stream)
{
    const float* feats = (const float*)d_in[0];  // [B,T,K] f32
    const float* trans = (const float*)d_in[1];  // [K,K]   f32
    const int*   tags  = (const int*)d_in[2];    // [B,T]   i32
    float* out = (float*)d_out;                  // [nll B][path_score B][paths B*T]

    float* logZ = (float*)d_ws;                                  // B floats
    unsigned char* bp = (unsigned char*)d_ws + 8192;             // B*T*K u8 (~25 MB)

    crf_forward_wmma<<<CRF_B / 16, 32, 0, stream>>>(feats, trans, logZ);
    crf_viterbi<<<CRF_B / 4, 256, 0, stream>>>(feats, trans, bp, out);
    crf_gold_nll<<<CRF_B, 128, 0, stream>>>(feats, trans, tags, logZ, out);
}
